// MultiHeadAttention_3650722202083
// MI455X (gfx1250) — matrix operand order repaired and checked by reading, not compiled, not run on hardware
//
#include <hip/hip_runtime.h>
#include <hip/hip_bf16.h>
#include <math.h>

#define DMODEL 1024
#define NHEADS 16
#define DKH    64
#define BATCH  8
#define SEQLEN 1024
#define BSROWS (BATCH * SEQLEN)   // 8192

typedef __attribute__((ext_vector_type(16))) _Float16     v16h;
typedef __attribute__((ext_vector_type(8)))  float        v8f;
typedef __attribute__((ext_vector_type(4)))  float        v4f;
typedef __attribute__((ext_vector_type(4)))  unsigned int v4u;

union FragH { v16h v; v4u u[2]; _Float16 h[16]; };
union H8    { v4u u; _Float16 h[8]; };

__device__ __forceinline__ v8f wmma16x16x32(const FragH& a, const FragH& b, v8f c) {
  // (neg_a, A, neg_b, B, c_mod, C, reuse_a, reuse_b)
  return __builtin_amdgcn_wmma_f32_16x16x32_f16(false, a.v, false, b.v, (short)0, c,
                                                false, false);
}

// ---------------------------------------------------------------------------
// NT GEMM: Y[m,n] = sum_k X[m,k] * W[n,k] + bias[n]
//   X: [M=8192, 1024] fp32 row-major, W: [1024, 1024] fp32 row-major.
// Block: 256 threads = 8 waves -> 64x128 tile; wave: 32x32 (2x2 WMMA accs).
// MODE 0: out f16 head layout [B,H,S,dk]
// MODE 1: out f16 transposed head layout [B,H,dk,S]
// MODE 2: out fp32 row-major [M, 1024]
// ---------------------------------------------------------------------------
#define BMT 64
#define BNT 128
#define BKT 32
#define LDH 40   // padded LDS row stride in halves (80 bytes, 16B multiple)

template<int MODE>
__global__ void __launch_bounds__(256)
gemm_nt_kernel(const float* __restrict__ X, const float* __restrict__ W,
               const float* __restrict__ bias, void* __restrict__ out)
{
  __shared__ _Float16 Ash[BMT * LDH];
  __shared__ _Float16 Bsh[BNT * LDH];

  const int tid  = threadIdx.x;
  const int lane = tid & 31;
  const int wave = tid >> 5;
  const int wr   = wave >> 2;          // 0..1  (M direction)
  const int wc   = wave & 3;           // 0..3  (N direction)
  const int half = lane >> 4;          // 0/1
  const int l15  = lane & 15;

  const int m0 = blockIdx.x * BMT;
  const int n0 = blockIdx.y * BNT;

  const int arow = tid >> 2;           // 0..63
  const int acol = (tid & 3) * 8;      // 0,8,16,24
  const int brow = tid >> 1;           // 0..127
  const int bcol = (tid & 1) * 16;     // 0,16

  const float* aptr = X + (size_t)(m0 + arow) * DMODEL + acol;
  const float* bptr = W + (size_t)(n0 + brow) * DMODEL + bcol;

  v8f c[2][2] = {};

  for (int k0 = 0; k0 < DMODEL; k0 += BKT) {
    // ---- stage A tile (fp32 -> f16 LDS) ----
    v4f a0 = *(const v4f*)(aptr + k0);
    v4f a1 = *(const v4f*)(aptr + k0 + 4);
    H8 ta;
#pragma unroll
    for (int i = 0; i < 4; i++) { ta.h[i] = (_Float16)a0[i]; ta.h[4 + i] = (_Float16)a1[i]; }
    *(v4u*)&Ash[arow * LDH + acol] = ta.u;

    // ---- stage B tile ----
    v4f b0 = *(const v4f*)(bptr + k0);
    v4f b1 = *(const v4f*)(bptr + k0 + 4);
    v4f b2 = *(const v4f*)(bptr + k0 + 8);
    v4f b3 = *(const v4f*)(bptr + k0 + 12);
    H8 tb0, tb1;
#pragma unroll
    for (int i = 0; i < 4; i++) {
      tb0.h[i] = (_Float16)b0[i]; tb0.h[4 + i] = (_Float16)b1[i];
      tb1.h[i] = (_Float16)b2[i]; tb1.h[4 + i] = (_Float16)b3[i];
    }
    *(v4u*)&Bsh[brow * LDH + bcol]     = tb0.u;
    *(v4u*)&Bsh[brow * LDH + bcol + 8] = tb1.u;

    if (k0 + BKT < DMODEL) {
      __builtin_prefetch(aptr + k0 + BKT, 0, 3);   // global_prefetch_b8
      __builtin_prefetch(bptr + k0 + BKT, 0, 3);
    }
    __syncthreads();

    // ---- fragments (A: 16 consecutive K; B: 16 consecutive K) ----
    FragH af[2], bf[2];
#pragma unroll
    for (int i = 0; i < 2; i++) {
      const _Float16* p = &Ash[(wr * 32 + i * 16 + l15) * LDH + half * 16];
      af[i].u[0] = *(const v4u*)p;
      af[i].u[1] = *(const v4u*)(p + 8);
    }
#pragma unroll
    for (int j = 0; j < 2; j++) {
      const _Float16* p = &Bsh[(wc * 32 + j * 16 + l15) * LDH + half * 16];
      bf[j].u[0] = *(const v4u*)p;
      bf[j].u[1] = *(const v4u*)(p + 8);
    }
#pragma unroll
    for (int i = 0; i < 2; i++)
#pragma unroll
      for (int j = 0; j < 2; j++)
        c[i][j] = wmma16x16x32(af[i], bf[j], c[i][j]);
    __syncthreads();
  }

  // ---- epilogue: C layout VGPR r, lanes 0-15 -> M=r, lanes 16-31 -> M=r+8; N=l15 ----
#pragma unroll
  for (int j = 0; j < 2; j++) {
    const int n = n0 + wc * 32 + j * 16 + l15;
    const float bval = bias[n];
#pragma unroll
    for (int i = 0; i < 2; i++) {
      const int mbase = m0 + wr * 32 + i * 16 + half * 8;
#pragma unroll
      for (int r = 0; r < 8; r++) {
        const int m = mbase + r;
        const float val = c[i][j][r] + bval;
        if constexpr (MODE == 2) {
          ((float*)out)[(size_t)m * DMODEL + n] = val;
        } else {
          const int bb = m >> 10, s = m & (SEQLEN - 1);
          const int hh = n >> 6,  d = n & (DKH - 1);
          _Float16* Y = (_Float16*)out;
          if constexpr (MODE == 0)
            Y[(((size_t)bb * NHEADS + hh) * SEQLEN + s) * DKH + d] = (_Float16)val;
          else
            Y[(((size_t)bb * NHEADS + hh) * DKH + d) * SEQLEN + s] = (_Float16)val;
        }
      }
    }
  }
}

// ---------------------------------------------------------------------------
// Pass 1: row sums of exp(Q K^T * scale). No stores of scores (recompute is
// ~free f16 WMMA work; Q/K heads are L2-resident). Writes 1/rowsum [B,H,S].
// One wave per 16-row query strip.
// ---------------------------------------------------------------------------
__global__ void __launch_bounds__(256)
rowsum_kernel(const _Float16* __restrict__ Qh, const _Float16* __restrict__ Kh,
              float* __restrict__ rinvBuf)
{
  const int tid  = threadIdx.x;
  const int lane = tid & 31;
  const int wave = tid >> 5;
  const int half = lane >> 4;
  const int l15  = lane & 15;

  const int bh = blockIdx.x;
  const int q0 = blockIdx.y * 128 + wave * 16;

  const _Float16* Qb = Qh + (size_t)bh * SEQLEN * DKH;
  const _Float16* Kb = Kh + (size_t)bh * SEQLEN * DKH;

  FragH qf[2];
#pragma unroll
  for (int kk = 0; kk < 2; kk++) {
    const _Float16* p = Qb + (size_t)(q0 + l15) * DKH + kk * 32 + half * 16;
    qf[kk].u[0] = *(const v4u*)p;
    qf[kk].u[1] = *(const v4u*)(p + 8);
  }

  float acc[8] = {0.f, 0.f, 0.f, 0.f, 0.f, 0.f, 0.f, 0.f};
  const float scale = 0.125f;   // 1/sqrt(64)

  for (int t = 0; t < SEQLEN / 16; t++) {
    v8f c = {};
#pragma unroll
    for (int kk = 0; kk < 2; kk++) {
      FragH kf;
      const _Float16* p = Kb + (size_t)(t * 16 + l15) * DKH + kk * 32 + half * 16;
      kf.u[0] = *(const v4u*)p;
      kf.u[1] = *(const v4u*)(p + 8);
      c = wmma16x16x32(qf[kk], kf, c);
    }
#pragma unroll
    for (int r = 0; r < 8; r++)
      acc[r] += __expf(c[r] * scale);
  }

  // total over 1024 columns: xor-reduce within each 16-lane half
#pragma unroll
  for (int r = 0; r < 8; r++) {
#pragma unroll
    for (int m = 1; m < 16; m <<= 1)
      acc[r] += __shfl_xor(acc[r], m, 32);
  }
  if (l15 == 0) {   // lane 0 -> rows 0..7, lane 16 -> rows 8..15
#pragma unroll
    for (int r = 0; r < 8; r++)
      rinvBuf[(size_t)bh * SEQLEN + q0 + half * 8 + r] = 1.0f / acc[r];
  }
}

// ---------------------------------------------------------------------------
// Pass 2: scores -> normalized attn (single write) fused with ctx = P @ V.
// Per kv-pair (32 cols): two 16x16 normalized E tiles are written to the attn
// output AND to a per-wave LDS buffer (f16), re-read as a 16x32 A-fragment
// (same-wave DS ops are in order), then 4 WMMAs against transposed-V frags.
// ---------------------------------------------------------------------------
__global__ void __launch_bounds__(256)
attn_ctx_kernel(const _Float16* __restrict__ Qh, const _Float16* __restrict__ Kh,
                const _Float16* __restrict__ Vt, const float* __restrict__ rinvBuf,
                float* __restrict__ attnOut, float* __restrict__ ctxOut)
{
  __shared__ _Float16 Esh[8][16 * LDH];   // per-wave 16 x (32+pad) f16 E tile

  const int tid  = threadIdx.x;
  const int lane = tid & 31;
  const int wave = tid >> 5;
  const int half = lane >> 4;
  const int l15  = lane & 15;

  const int bh = blockIdx.x;                    // b*H + h
  const int b  = bh >> 4;
  const int h  = bh & (NHEADS - 1);
  const int q0 = blockIdx.y * 128 + wave * 16;

  const _Float16* Qb = Qh + (size_t)bh * SEQLEN * DKH;
  const _Float16* Kb = Kh + (size_t)bh * SEQLEN * DKH;
  const _Float16* Vb = Vt + (size_t)bh * DKH * SEQLEN;
  float* P = attnOut + ((size_t)bh * SEQLEN + q0) * SEQLEN;  // strip [16, SEQ]
  _Float16* Em = &Esh[wave][0];

  FragH qf[2];
#pragma unroll
  for (int kk = 0; kk < 2; kk++) {
    const _Float16* p = Qb + (size_t)(q0 + l15) * DKH + kk * 32 + half * 16;
    qf[kk].u[0] = *(const v4u*)p;
    qf[kk].u[1] = *(const v4u*)(p + 8);
  }

  // 1/rowsum for this lane's C-layout rows {half*8 + r}
  const float* rp = rinvBuf + (size_t)bh * SEQLEN + q0 + half * 8;
  v4f rv0 = *(const v4f*)rp;
  v4f rv1 = *(const v4f*)(rp + 4);
  float rsv[8];
#pragma unroll
  for (int i = 0; i < 4; i++) { rsv[i] = rv0[i]; rsv[4 + i] = rv1[i]; }

  const float scale = 0.125f;
  v8f cc[4] = {};

  for (int pair = 0; pair < SEQLEN / 32; pair++) {
#pragma unroll
    for (int sub = 0; sub < 2; sub++) {
      const int t = pair * 2 + sub;
      v8f c = {};
#pragma unroll
      for (int kk = 0; kk < 2; kk++) {
        FragH kf;
        const _Float16* p = Kb + (size_t)(t * 16 + l15) * DKH + kk * 32 + half * 16;
        kf.u[0] = *(const v4u*)p;
        kf.u[1] = *(const v4u*)(p + 8);
        c = wmma16x16x32(qf[kk], kf, c);
      }
#pragma unroll
      for (int r = 0; r < 8; r++) {
        const float e = __expf(c[r] * scale) * rsv[r];   // normalized prob
        P[(size_t)(half * 8 + r) * SEQLEN + t * 16 + l15] = e;     // final attn
        Em[(half * 8 + r) * LDH + sub * 16 + l15] = (_Float16)e;   // for P@V
      }
    }

    // A-fragment of the 16x32 normalized tile from LDS (same-wave, in order)
    FragH af;
    const _Float16* ap = &Em[l15 * LDH + half * 16];
    af.u[0] = *(const v4u*)ap;
    af.u[1] = *(const v4u*)(ap + 8);

#pragma unroll
    for (int j = 0; j < 4; j++) {
      FragH vf;
      const _Float16* vp = Vb + (size_t)(j * 16 + l15) * SEQLEN + pair * 32 + half * 16;
      vf.u[0] = *(const v4u*)vp;
      vf.u[1] = *(const v4u*)(vp + 8);
      cc[j] = wmma16x16x32(af, vf, cc[j]);
    }
  }

  // ctx -> concat layout [B, S, D]: column d_model = h*64 + j*16 + l15
#pragma unroll
  for (int j = 0; j < 4; j++) {
    const int d = h * DKH + j * 16 + l15;
#pragma unroll
    for (int r = 0; r < 8; r++) {
      const int s = q0 + half * 8 + r;
      ctxOut[((size_t)b * SEQLEN + s) * DMODEL + d] = cc[j][r];
    }
  }
}

// ---------------------------------------------------------------------------
extern "C" void kernel_launch(void* const* d_in, const int* in_sizes, int n_in,
                              void* d_out, int out_size, void* d_ws, size_t ws_size,
                              hipStream_t stream)
{
  const float* q   = (const float*)d_in[0];
  const float* k   = (const float*)d_in[1];
  const float* v   = (const float*)d_in[2];
  const float* Wq  = (const float*)d_in[3];
  const float* bq  = (const float*)d_in[4];
  const float* Wk  = (const float*)d_in[5];
  const float* bk  = (const float*)d_in[6];
  const float* Wv  = (const float*)d_in[7];
  const float* bvv = (const float*)d_in[8];
  const float* Wo  = (const float*)d_in[9];
  const float* bo  = (const float*)d_in[10];

  float* out  = (float*)d_out;                       // [B*S, D] = [8192, 1024]
  float* attn = out + (size_t)BSROWS * DMODEL;       // [B, H, S, S]

  const size_t headElems = (size_t)BATCH * NHEADS * SEQLEN * DKH;  // 8.4M
  char* ws = (char*)d_ws;
  _Float16* Qh = (_Float16*)ws;  ws += headElems * sizeof(_Float16);
  _Float16* Kh = (_Float16*)ws;  ws += headElems * sizeof(_Float16);
  _Float16* Vt = (_Float16*)ws;  ws += headElems * sizeof(_Float16);
  float*    ctx = (float*)ws;    ws += (size_t)BSROWS * DMODEL * sizeof(float);
  float*    rinv = (float*)ws;   // [B*H*S] = 131072 floats

  dim3 blk(256);
  dim3 gg(BSROWS / BMT, DMODEL / BNT);   // 128 x 8
  gemm_nt_kernel<0><<<gg, blk, 0, stream>>>(q, Wq, bq, (void*)Qh);
  gemm_nt_kernel<0><<<gg, blk, 0, stream>>>(k, Wk, bk, (void*)Kh);
  gemm_nt_kernel<1><<<gg, blk, 0, stream>>>(v, Wv, bvv, (void*)Vt);

  dim3 ga(BATCH * NHEADS, SEQLEN / 128); // 128 x 8
  rowsum_kernel<<<ga, blk, 0, stream>>>(Qh, Kh, rinv);
  attn_ctx_kernel<<<ga, blk, 0, stream>>>(Qh, Kh, Vt, rinv, attn, ctx);

  gemm_nt_kernel<2><<<gg, blk, 0, stream>>>(ctx, Wo, bo, (void*)out);
}
